// Local_update_Layer_32822140076359
// MI455X (gfx1250) — matrix operand order repaired and checked by reading, not compiled, not run on hardware
//
#include <hip/hip_runtime.h>
#include <hip/hip_bf16.h>
#include <cstddef>

// ---------------------------------------------------------------------------
// CDNA5 (gfx1250) wave32 WMMA types & helpers
// ---------------------------------------------------------------------------
typedef __attribute__((ext_vector_type(16))) __bf16 v16bf;
typedef __attribute__((ext_vector_type(8)))  __bf16 v8bf;   // 16 B -> ds_load_b128
typedef __attribute__((ext_vector_type(4)))  __bf16 v4bf;   // 8 B  -> ds_store_b64
typedef __attribute__((ext_vector_type(8)))  float  v8f;
typedef __attribute__((ext_vector_type(4)))  float  v4f;    // 16 B -> global_load_b128

#define DH   64          // hidden dim
#define K1P  160         // 129 padded up to 5*32
#define NN   50000
#define EE   800000

__device__ __forceinline__ v8f wmma_bf16(v16bf a, v16bf b, v8f c) {
  // D = A(16x32 bf16) * B(32x16 bf16) + C(16x16 f32)
  return __builtin_amdgcn_wmma_f32_16x16x32_bf16(
      false, a, false, b, (short)0, c, false, false);
}

// A fragment (16x32, MxK) from a row-major bf16 tile T[m][k], row stride ld
// (ld multiple of 8).  Per-lane data is two 16-byte contiguous runs:
//   lanes 0-15 : row M=lane,    K = kb + {0..7} and kb + {8..15}
//   lanes16-31 : row M=lane-16, K = kb + {16..23} and kb + {24..31}
__device__ __forceinline__ v16bf load_a_frag(const __bf16* T, int ld, int kb, int lane) {
  int m     = lane & 15;
  int khalf = (lane >> 4) << 4;   // 0 or 16
  const __bf16* base = T + m * ld + kb + khalf;
  union { v16bf v; v8bf h[2]; } u;
  u.h[0] = *(const v8bf*)(base);        // ds_load_b128
  u.h[1] = *(const v8bf*)(base + 8);    // ds_load_b128
  return u.v;
}

// B fragment from a *fragment-major packed* weight buffer: block blk holds a
// 32x16 (KxN) tile as [lane][16 contiguous bf16].
__device__ __forceinline__ v16bf load_b_packed(const __bf16* Wp, int blk, int lane) {
  const __bf16* base = Wp + (((blk << 5) + lane) << 4);
  union { v16bf v; v8bf h[2]; } u;
  u.h[0] = *(const v8bf*)(base);        // ds_load_b128
  u.h[1] = *(const v8bf*)(base + 8);    // ds_load_b128
  return u.v;
}

// Pack a global fp32 weight (row stride Wld, valid Kvalid x Nvalid) into
// fragment-major bf16 blocks of 32x16 covering K x N (zero padded).
// Packed element p: j=p&15, lane=(p>>4)&31, blk=p>>9; blk=(ks*nNT+nt);
//   k = ks*32 + ((lane>>4)<<4) + j , n = nt*16 + (lane&15)
__device__ __forceinline__ void pack_b(__bf16* dst, const float* W,
                                       int K, int N, int Kvalid, int Nvalid,
                                       int Wld, int tid, int nthreads) {
  int nNT = N >> 4;
  int total = K * N;
  for (int p = tid; p < total; p += nthreads) {
    int j   = p & 15;
    int l   = (p >> 4) & 31;
    int blk = p >> 9;
    int ks  = blk / nNT, nt = blk % nNT;
    int k   = ks * 32 + ((l >> 4) << 4) + j;
    int n   = nt * 16 + (l & 15);
    dst[p] = (k < Kvalid && n < Nvalid) ? (__bf16)W[k * Wld + n] : (__bf16)0.0f;
  }
}

// C/D element (m,n): lane l, vgpr i -> M = i + 8*(l>=16), N = (l&15) + 16*nt.
// Bias + SiLU on 4 N-tiles, store bf16 [16][ld] tile (column-strided stores).
__device__ __forceinline__ void silu_bias_store(const v8f* acc, const float* bias,
                                                __bf16* T, int ld, int lane) {
  int ncol = lane & 15;
  int mrow = (lane >> 4) << 3;
#pragma unroll
  for (int nt = 0; nt < 4; ++nt) {
    int n = nt * 16 + ncol;
    float bv = bias[n];
#pragma unroll
    for (int i = 0; i < 8; ++i) {
      float h = acc[nt][i] + bv;
      h = h / (1.0f + __expf(-h));          // SiLU
      T[(mrow + i) * ld + n] = (__bf16)h;
    }
  }
}

// ---------------------------------------------------------------------------
// Kernel 0: zero workspace (sums[N*192] ++ deg[N])
// ---------------------------------------------------------------------------
__global__ void zero_ws(float* __restrict__ p, size_t n) {
  size_t i = (size_t)blockIdx.x * blockDim.x + threadIdx.x;
  size_t stride = (size_t)gridDim.x * blockDim.x;
  for (; i < n; i += stride) p[i] = 0.0f;
}

// ---------------------------------------------------------------------------
// Kernel 1: edge messages.  4 waves / block, one 16-edge tile per wave.
//   K layout (permuted vs reference, weights permuted to match):
//     k 0..127 = [xh[i0], xh[i1]],  k 128 = |r|^2,  k 129..159 = zero pad
// ---------------------------------------------------------------------------
__global__ __launch_bounds__(128) void edge_msg(
    const float* __restrict__ x, const float* __restrict__ xh,
    const int* __restrict__ e,
    const float* __restrict__ sW1, const float* __restrict__ sb1,
    const float* __restrict__ sW2, const float* __restrict__ sb2,
    float* __restrict__ sums, float* __restrict__ deg, int numTiles) {
  __shared__ __bf16 wB1[K1P * DH];          // 20 KB, fragment-major packed
  __shared__ __bf16 wB2[DH * DH];           //  8 KB, fragment-major packed
  __shared__ float  lb1[DH], lb2[DH];
  __shared__ __bf16 stage[4][16 * K1P];     // 20 KB per-wave activation tiles
  __shared__ float  sDir[4][16][3];
  __shared__ int    sI0[4][16], sI1[4][16];

  const int tid = threadIdx.x;

  // Pack sW1 with K-permutation: packed k<128 -> orig k+1; k==128 -> orig 0.
  for (int p = tid; p < K1P * DH; p += 128) {
    int j = p & 15, l = (p >> 4) & 31, blk = p >> 9;
    int ks = blk >> 2, nt = blk & 3;                 // nNT = 4
    int k = ks * 32 + ((l >> 4) << 4) + j;           // permuted k, 0..159
    int n = nt * 16 + (l & 15);
    int korig = (k < 128) ? (k + 1) : ((k == 128) ? 0 : -1);
    wB1[p] = (korig >= 0) ? (__bf16)sW1[korig * DH + n] : (__bf16)0.0f;
  }
  pack_b(wB2, sW2, DH, DH, DH, DH, DH, tid, 128);
  if (tid < DH) { lb1[tid] = sb1[tid]; lb2[tid] = sb2[tid]; }
  __syncthreads();

  const int w = tid >> 5, lane = tid & 31;
  __bf16* st = stage[w];

  // Zero-pad K columns 129..159 once (never overwritten below).
  if (lane < 31) {
#pragma unroll
    for (int m = 0; m < 16; ++m) st[m * K1P + 129 + lane] = (__bf16)0.0f;
  }

  for (int tile = blockIdx.x * 4 + w; tile < numTiles; tile += gridDim.x * 4) {
    const int eb = tile * 16;

    if (tile + gridDim.x * 4 < numTiles)
      __builtin_prefetch(e + (tile + gridDim.x * 4) * 16, 0, 1);  // global_prefetch

    // Geometry: lanes 0..15 each own one edge.
    if (lane < 16) {
      int ei = eb + lane;
      int i0 = e[ei], i1 = e[EE + ei];
      float rx = x[i0 * 3 + 0] - x[i1 * 3 + 0];
      float ry = x[i0 * 3 + 1] - x[i1 * 3 + 1];
      float rz = x[i0 * 3 + 2] - x[i1 * 3 + 2];
      float d2  = rx * rx + ry * ry + rz * rz;
      float inv = rsqrtf(d2);
      sDir[w][lane][0] = rx * inv;
      sDir[w][lane][1] = ry * inv;
      sDir[w][lane][2] = rz * inv;
      sI0[w][lane] = i0;
      sI1[w][lane] = i1;
      st[lane * K1P + 128] = (__bf16)d2;
    }

    // Feature gather: 16 edges x 32 quads; b128 global load -> b64 LDS store.
    for (int t = lane; t < 16 * 32; t += 32) {
      int m = t >> 5, q = t & 31;
      int j = q << 2;                                  // 0,4,...,124
      int node = (j < 64) ? sI0[w][m] : sI1[w][m];
      v4f f = *(const v4f*)(xh + (size_t)node * DH + (j & 63));
      v4bf o;
      o[0] = (__bf16)f[0]; o[1] = (__bf16)f[1];
      o[2] = (__bf16)f[2]; o[3] = (__bf16)f[3];
      *(v4bf*)(st + m * K1P + j) = o;
    }

    // GEMM1: [16x160] @ [160x64]  (5 K-steps x 4 N-tiles)
    v8f acc[4] = {};
#pragma unroll
    for (int ks = 0; ks < 5; ++ks) {
      v16bf a = load_a_frag(st, K1P, ks * 32, lane);
#pragma unroll
      for (int nt = 0; nt < 4; ++nt)
        acc[nt] = wmma_bf16(a, load_b_packed(wB1, ks * 4 + nt, lane), acc[nt]);
    }
    silu_bias_store(acc, lb1, st, K1P, lane);     // reuse cols 0..63 of stage

    // GEMM2: [16x64] @ [64x64]
    v8f acc2[4] = {};
#pragma unroll
    for (int ks = 0; ks < 2; ++ks) {
      v16bf a = load_a_frag(st, K1P, ks * 32, lane);
#pragma unroll
      for (int nt = 0; nt < 4; ++nt)
        acc2[nt] = wmma_bf16(a, load_b_packed(wB2, ks * 4 + nt, lane), acc2[nt]);
    }

    // Scatter mij (x) dirv onto destination nodes (L2-side f32 atomics).
    const int ncol = lane & 15, mrow = (lane >> 4) << 3;
#pragma unroll
    for (int nt = 0; nt < 4; ++nt) {
      int n = nt * 16 + ncol;
      float bv = lb2[n];
#pragma unroll
      for (int i = 0; i < 8; ++i) {
        int m = mrow + i;
        float mij = acc2[nt][i] + bv;
        float* dst = sums + (size_t)sI0[w][m] * 192 + n * 3;
        atomicAdd(dst + 0, mij * sDir[w][m][0]);
        atomicAdd(dst + 1, mij * sDir[w][m][1]);
        atomicAdd(dst + 2, mij * sDir[w][m][2]);
      }
    }
    if (lane < 16) atomicAdd(deg + sI0[w][lane], 1.0f);
  }
}

// ---------------------------------------------------------------------------
// Kernel 2: node finalize. 2 waves / block, one 16-node tile per wave.
// Order: aggregate -> n2 -> n1 -> x_out -> n3 (lets bufT be reused 3x).
// ---------------------------------------------------------------------------
__global__ __launch_bounds__(64) void node_fin(
    const float* __restrict__ xh,
    const float* __restrict__ sums, const float* __restrict__ deg,
    const float* __restrict__ n1W1, const float* __restrict__ n1b1,
    const float* __restrict__ n1W2, const float* __restrict__ n1b2,
    const float* __restrict__ n2W1, const float* __restrict__ n2b1,
    const float* __restrict__ n2W2, const float* __restrict__ n2b2,
    const float* __restrict__ n3W1, const float* __restrict__ n3b1,
    const float* __restrict__ n3W2, const float* __restrict__ n3b2,
    float* __restrict__ out_x, float* __restrict__ out_h,
    float* __restrict__ out_nrm, int numTiles) {
  __shared__ __bf16 w11[DH * DH];           //  8 KB packed
  __shared__ __bf16 w21[DH * DH];           //  8 KB packed
  __shared__ __bf16 w31[2 * DH * DH];       // 16 KB packed (K=128)
  __shared__ __bf16 w32[DH * DH];           //  8 KB packed
  __shared__ __bf16 w12p[DH * 16];          //  2 KB packed (64x3 -> 64x16)
  __shared__ __bf16 w22p[DH * 16];          //  2 KB packed (64x1 -> 64x16)
  __shared__ float  bb11[DH], bb21[DH], bb31[DH], bb32[DH], bb12[4], bb22[1];
  __shared__ __bf16 bufY[2][16 * DH];       // per-wave intermediate
  __shared__ __bf16 bufT[2][48 * DH];       // n2 in (48x64) / xh (16x64) / n3 in (16x128)
  __shared__ float  sXo[2][16][3];

  const int tid = threadIdx.x;
  pack_b(w11, n1W1, DH, DH, DH, DH, DH, tid, 64);
  pack_b(w21, n2W1, DH, DH, DH, DH, DH, tid, 64);
  pack_b(w31, n3W1, 2 * DH, DH, 2 * DH, DH, DH, tid, 64);
  pack_b(w32, n3W2, DH, DH, DH, DH, DH, tid, 64);
  pack_b(w12p, n1W2, DH, 16, DH, 3, 3, tid, 64);
  pack_b(w22p, n2W2, DH, 16, DH, 1, 1, tid, 64);
  if (tid < DH) { bb11[tid] = n1b1[tid]; bb21[tid] = n2b1[tid];
                  bb31[tid] = n3b1[tid]; bb32[tid] = n3b2[tid]; }
  if (tid < 3)  bb12[tid] = n1b2[tid];
  if (tid == 0) bb22[0] = n2b2[0];
  __syncthreads();

  const int w = tid >> 5, lane = tid & 31;
  const int ncol = lane & 15, mrow = (lane >> 4) << 3;

  for (int tile = blockIdx.x * 2 + w; tile < numTiles; tile += gridDim.x * 2) {
    const int nb0 = tile * 16;

    // (a) scatter-mean + per-channel 3-vector norm; stage n2 input [48x64].
    for (int t = lane; t < 16 * 64; t += 32) {
      int m = t >> 6, d = t & 63;
      int node = nb0 + m;
      float invd = 1.0f / fmaxf(deg[node], 1.0f);
      const float* s = sums + (size_t)node * 192 + d * 3;
      float v0 = s[0] * invd, v1 = s[1] * invd, v2 = s[2] * invd;
      float nr = sqrtf(v0 * v0 + v1 * v1 + v2 * v2);
      out_nrm[(size_t)node * DH + d] = nr;
      bufT[w][(3 * m + 0) * DH + d] = (__bf16)v0;
      bufT[w][(3 * m + 1) * DH + d] = (__bf16)v1;
      bufT[w][(3 * m + 2) * DH + d] = (__bf16)v2;
    }

    // (b) n2: rows are (node, axis); 3 M-tiles. 64 -> SiLU -> 1 (N padded 16).
#pragma unroll
    for (int mt = 0; mt < 3; ++mt) {
      v8f a2[4] = {};
#pragma unroll
      for (int ks = 0; ks < 2; ++ks) {
        v16bf a = load_a_frag(bufT[w] + mt * 16 * DH, DH, ks * 32, lane);
#pragma unroll
        for (int nt = 0; nt < 4; ++nt)
          a2[nt] = wmma_bf16(a, load_b_packed(w21, ks * 4 + nt, lane), a2[nt]);
      }
      silu_bias_store(a2, bb21, bufY[w], DH, lane);
      v8f r = {};
#pragma unroll
      for (int ks = 0; ks < 2; ++ks)
        r = wmma_bf16(load_a_frag(bufY[w], DH, ks * 32, lane),
                      load_b_packed(w22p, ks, lane), r);
      if (ncol == 0) {                       // only column 0 is real output
#pragma unroll
        for (int i = 0; i < 8; ++i) {
          int row = mt * 16 + mrow + i;      // row = 3*m + axis
          sXo[w][row / 3][row % 3] = r[i];
        }
      }
    }

    // (c) n1: stage xh tile (b128 load / b64 store), 64 -> SiLU -> 3 (padded).
    for (int t = lane; t < 16 * 16; t += 32) {
      int m = t >> 4, j = (t & 15) << 2;
      v4f f = *(const v4f*)(xh + (size_t)(nb0 + m) * DH + j);
      v4bf o;
      o[0] = (__bf16)f[0]; o[1] = (__bf16)f[1];
      o[2] = (__bf16)f[2]; o[3] = (__bf16)f[3];
      *(v4bf*)(bufT[w] + m * DH + j) = o;
    }
    v8f a1[4] = {};
#pragma unroll
    for (int ks = 0; ks < 2; ++ks) {
      v16bf a = load_a_frag(bufT[w], DH, ks * 32, lane);
#pragma unroll
      for (int nt = 0; nt < 4; ++nt)
        a1[nt] = wmma_bf16(a, load_b_packed(w11, ks * 4 + nt, lane), a1[nt]);
    }
    silu_bias_store(a1, bb11, bufY[w], DH, lane);
    v8f x1 = {};
#pragma unroll
    for (int ks = 0; ks < 2; ++ks)
      x1 = wmma_bf16(load_a_frag(bufY[w], DH, ks * 32, lane),
                     load_b_packed(w12p, ks, lane), x1);

    // (d) x_out = n1(xh) + n2(v^T)
    if (ncol < 3) {
#pragma unroll
      for (int i = 0; i < 8; ++i) {
        int m = mrow + i;
        out_x[(size_t)(nb0 + m) * 3 + ncol] =
            x1[i] + bb12[ncol] + sXo[w][m][ncol] + bb22[0];
      }
    }

    // (e) n3: concat(xh, nrm) [16x128] -> 64 -> SiLU -> 64
    for (int t = lane; t < 16 * 32; t += 32) {
      int m = t >> 5, j = (t & 31) << 2;
      int node = nb0 + m;
      const float* src = (j < 64) ? xh + (size_t)node * DH + j
                                  : out_nrm + (size_t)node * DH + (j - 64);
      v4f f = *(const v4f*)src;
      v4bf o;
      o[0] = (__bf16)f[0]; o[1] = (__bf16)f[1];
      o[2] = (__bf16)f[2]; o[3] = (__bf16)f[3];
      *(v4bf*)(bufT[w] + m * 128 + j) = o;   // reuse bufT as [16][128]
    }
    v8f a3[4] = {};
#pragma unroll
    for (int ks = 0; ks < 4; ++ks) {
      v16bf a = load_a_frag(bufT[w], 128, ks * 32, lane);
#pragma unroll
      for (int nt = 0; nt < 4; ++nt)
        a3[nt] = wmma_bf16(a, load_b_packed(w31, ks * 4 + nt, lane), a3[nt]);
    }
    silu_bias_store(a3, bb31, bufY[w], DH, lane);
    v8f h2[4] = {};
#pragma unroll
    for (int ks = 0; ks < 2; ++ks) {
      v16bf a = load_a_frag(bufY[w], DH, ks * 32, lane);
#pragma unroll
      for (int nt = 0; nt < 4; ++nt)
        h2[nt] = wmma_bf16(a, load_b_packed(w32, ks * 4 + nt, lane), h2[nt]);
    }
#pragma unroll
    for (int nt = 0; nt < 4; ++nt) {
      int n = nt * 16 + ncol;
      float bv = bb32[n];
#pragma unroll
      for (int i = 0; i < 8; ++i)
        out_h[(size_t)(nb0 + mrow + i) * DH + n] = h2[nt][i] + bv;
    }
  }
}

// ---------------------------------------------------------------------------
// Host entry
// ---------------------------------------------------------------------------
extern "C" void kernel_launch(void* const* d_in, const int* in_sizes, int n_in,
                              void* d_out, int out_size, void* d_ws, size_t ws_size,
                              hipStream_t stream) {
  const float* x    = (const float*)d_in[0];
  const float* xh   = (const float*)d_in[1];
  const int*   e    = (const int*)d_in[2];
  const float* sW1  = (const float*)d_in[3];
  const float* sb1  = (const float*)d_in[4];
  const float* sW2  = (const float*)d_in[5];
  const float* sb2  = (const float*)d_in[6];
  const float* n1W1 = (const float*)d_in[7];
  const float* n1b1 = (const float*)d_in[8];
  const float* n1W2 = (const float*)d_in[9];
  const float* n1b2 = (const float*)d_in[10];
  const float* n2W1 = (const float*)d_in[11];
  const float* n2b1 = (const float*)d_in[12];
  const float* n2W2 = (const float*)d_in[13];
  const float* n2b2 = (const float*)d_in[14];
  const float* n3W1 = (const float*)d_in[15];
  const float* n3b1 = (const float*)d_in[16];
  const float* n3W2 = (const float*)d_in[17];
  const float* n3b2 = (const float*)d_in[18];

  float* sums = (float*)d_ws;                    // [N * 64 * 3]
  float* deg  = sums + (size_t)NN * 192;         // [N]

  float* out     = (float*)d_out;
  float* out_x   = out;                          // [N*3]
  float* out_h   = out + (size_t)3 * NN;         // [N*64]
  float* out_nrm = out + (size_t)67 * NN;        // [N*64]

  zero_ws<<<1024, 256, 0, stream>>>(sums, (size_t)NN * 193);

  edge_msg<<<1250, 128, 0, stream>>>(x, xh, e, sW1, sb1, sW2, sb2,
                                     sums, deg, EE / 16);

  node_fin<<<512, 64, 0, stream>>>(xh, sums, deg,
                                   n1W1, n1b1, n1W2, n1b2,
                                   n2W1, n2b1, n2W2, n2b2,
                                   n3W1, n3b1, n3W2, n3b2,
                                   out_x, out_h, out_nrm, NN / 16);
}
